// CrossMultiHeadAttention_28312424415313
// MI455X (gfx1250) — matrix operand order repaired and checked by reading, not compiled, not run on hardware
//
#include <hip/hip_runtime.h>
#include <hip/hip_bf16.h>
#include <stdint.h>

typedef __attribute__((ext_vector_type(16))) __bf16 v16bf;
typedef __attribute__((ext_vector_type(8)))  __bf16 v8bf;
typedef __attribute__((ext_vector_type(8)))  float  v8f;
typedef __attribute__((ext_vector_type(4)))  int    v4i;

// fixed problem shape
#define NSEQ 2048
#define RTOT 8192      // B*N
#define NHEAD 4
#define DHEAD 64
#define LDT 48         // LDS row stride (elems) = 96B, multiple of 16B

// -------- CDNA5 async global->LDS path (guarded; falls back to vector copy) ----
#if defined(__HIP_DEVICE_COMPILE__) && \
    __has_builtin(__builtin_amdgcn_global_load_async_to_lds_b128) && \
    __has_builtin(__builtin_amdgcn_s_wait_asynccnt)
#define USE_ASYNC 1
#else
#define USE_ASYNC 0
#endif

__device__ __forceinline__ void cp16_lds(const __bf16* g, __bf16* l) {
#if USE_ASYNC
  __builtin_amdgcn_global_load_async_to_lds_b128(
      (__attribute__((address_space(1))) v4i*)g,
      (__attribute__((address_space(3))) v4i*)l, 0, 0);
#else
  *(v8bf*)l = *(const v8bf*)g;
#endif
}

__device__ __forceinline__ void async_fence() {
#if USE_ASYNC
  __builtin_amdgcn_s_wait_asynccnt(0);
#endif
}

// ---------------- helpers ----------------

__device__ __forceinline__ v8f vzero8() {
  v8f z;
#pragma unroll
  for (int i = 0; i < 8; ++i) z[i] = 0.0f;
  return z;
}

// A-fragment (16x32 bf16): per lane, elems 0-7 at k-base, 8-15 at k-base+16
__device__ __forceinline__ v16bf frag_split(const __bf16* lo, const __bf16* hi) {
  v8bf l = *(const v8bf*)lo;
  v8bf h = *(const v8bf*)hi;
  v16bf r;
#pragma unroll
  for (int i = 0; i < 8; ++i) { r[i] = l[i]; r[i + 8] = h[i]; }
  return r;
}

// B-fragment (32x16 bf16): per lane, elems 0-7 at k-base, 8-15 at k-base+16
__device__ __forceinline__ v16bf frag_contig(const __bf16* p) {
  return frag_split(p, p + 16);
}

__device__ __forceinline__ v8f wmma_bf16(v16bf a, v16bf b, v8f c) {
  return __builtin_amdgcn_wmma_f32_16x16x32_bf16(false, a, false, b,
                                                 (short)0, c, false, false);
}

// ---------------- fp32 -> bf16 bulk convert (native cast) ----------------

__global__ __launch_bounds__(128)
void cvt_f32_bf16(const float* __restrict__ src, __bf16* __restrict__ dst) {
  size_t i = ((size_t)blockIdx.x * 128 + threadIdx.x) * 8;
  float4 a = *(const float4*)(src + i);
  float4 b = *(const float4*)(src + i + 4);
  v8bf o;
  o[0] = (__bf16)a.x; o[1] = (__bf16)a.y; o[2] = (__bf16)a.z; o[3] = (__bf16)a.w;
  o[4] = (__bf16)b.x; o[5] = (__bf16)b.y; o[6] = (__bf16)b.z; o[7] = (__bf16)b.w;
  *(v8bf*)(dst + i) = o;
}

// ---------------- bf16 GEMM:  out = [A0|A1] @ W^T + bias (+resid) ----------------
// A0/A1, W already bf16. BM=BN=64, BK=32, 4 waves/block.
// headMode==1: bf16 out, head layout [B,H,N,DH], scaled by scl.
// headMode==0 && outH: bf16 out, plain [R,O].
// outH==nullptr: fp32 out [R,O], optional residual.

__global__ __launch_bounds__(128)
void gemm_bf16_wmma(const __bf16* __restrict__ A0, const __bf16* __restrict__ A1,
                    int Ksplit,
                    const __bf16* __restrict__ W, const float* __restrict__ bias,
                    const float* __restrict__ resid,
                    float* __restrict__ outF, __bf16* __restrict__ outH,
                    int headMode, int K, int O, float scl) {
  __shared__ __align__(16) __bf16 As[64 * LDT];
  __shared__ __align__(16) __bf16 Ws[64 * LDT];
  const int tid = threadIdx.x;
  const int lane = tid & 31, wv = tid >> 5;
  const int rowBase = blockIdx.x * 64;
  const int colBase = blockIdx.y * 64;
  const int K1 = K - Ksplit;
  const int nloc = lane & 15;

  v8f acc[4];
#pragma unroll
  for (int i = 0; i < 4; ++i) acc[i] = vzero8();

  for (int k0 = 0; k0 < K; k0 += 32) {
    __syncthreads();
    // prefetch next k-tile of W (global_prefetch_b8)
    if (k0 + 32 < K)
      __builtin_prefetch(W + (size_t)(colBase + (tid & 63)) * K + k0 + 32, 0, 0);
    // A tile: 64 rows x 32 k, 16B chunks
    for (int u = tid; u < 64 * 4; u += 128) {
      int row = u >> 2, c = (u & 3) * 8;
      int gk = k0 + c;
      const __bf16* src = (gk < Ksplit)
          ? (A0 + (size_t)(rowBase + row) * Ksplit + gk)
          : (A1 + (size_t)(rowBase + row) * K1 + (gk - Ksplit));
      cp16_lds(src, &As[row * LDT + c]);
    }
    // W tile: 64 out-cols x 32 k
    for (int u = tid; u < 64 * 4; u += 128) {
      int row = u >> 2, c = (u & 3) * 8;
      cp16_lds(W + (size_t)(colBase + row) * K + (k0 + c), &Ws[row * LDT + c]);
    }
    async_fence();
    __syncthreads();

    const int m = wv * 16 + nloc;
    const int ka = (lane >> 4) * 8;
    v16bf a = frag_split(&As[m * LDT + ka], &As[m * LDT + ka + 16]);
    const int kb = (lane >> 4) * 8;
#pragma unroll
    for (int nt = 0; nt < 4; ++nt) {
      v16bf b = frag_contig(&Ws[(nt * 16 + nloc) * LDT + kb]);
      acc[nt] = wmma_bf16(a, b, acc[nt]);
    }
  }

  const int mb = (lane >> 4) * 8;
#pragma unroll
  for (int nt = 0; nt < 4; ++nt) {
    int col = colBase + nt * 16 + nloc;
    float bv = bias ? bias[col] : 0.0f;
#pragma unroll
    for (int v = 0; v < 8; ++v) {
      int r = rowBase + wv * 16 + mb + v;
      float val = acc[nt][v] + bv;
      if (outH) {
        val *= scl;
        if (headMode) {  // [B,H,N,DH], all powers of two -> shifts
          size_t idx = ((((size_t)(r >> 11) * NHEAD + (col >> 6)) * NSEQ +
                         (r & (NSEQ - 1))) * DHEAD) + (col & (DHEAD - 1));
          outH[idx] = (__bf16)val;
        } else {
          outH[(size_t)r * O + col] = (__bf16)val;
        }
      } else {
        if (resid) val += resid[(size_t)r * O + col];
        outF[(size_t)r * O + col] = val;
      }
    }
  }
}

// ---------------- flash cross-attention (one direction) ----------------
// Q,K,V: bf16 [B,H,N,64].  out: bf16 merged [B,Nq,256] at head column h*64.

__global__ __launch_bounds__(128)
void attn_wmma(const __bf16* __restrict__ Q, const __bf16* __restrict__ Kd,
               const __bf16* __restrict__ V, __bf16* __restrict__ outM,
               int Nq, int Nk) {
  const int tid = threadIdx.x, lane = tid & 31, wv = tid >> 5;
  const int qTile = blockIdx.x, h = blockIdx.y, b = blockIdx.z;
  const size_t bh = (size_t)(b * NHEAD + h);
  const __bf16* Qbh = Q + bh * (size_t)Nq * DHEAD;
  const __bf16* Kbh = Kd + bh * (size_t)Nk * DHEAD;
  const __bf16* Vbh = V + bh * (size_t)Nk * DHEAD;

  __shared__ __align__(16) __bf16 Ks[64 * 72];      // [key][d]
  __shared__ __align__(16) __bf16 Vs[64 * 72];      // [d][key] (transposed)
  __shared__ __align__(16) __bf16 Ps[4][16 * 72];   // per-wave P [row][key]

  const int nloc = lane & 15;
  const int mb = (lane >> 4) * 8;
  const int kh = (lane >> 4) * 8;

  // Q fragments for this wave's 16 rows, two 32-wide k windows (direct global)
  v16bf qa0, qa1;
  {
    int m = qTile * 64 + wv * 16 + nloc;
    int ka = (lane >> 4) * 8;
    const __bf16* qr = Qbh + (size_t)m * DHEAD;
    qa0 = frag_split(qr + ka, qr + ka + 16);
    qa1 = frag_split(qr + 32 + ka, qr + 32 + ka + 16);
  }

  v8f oacc[4];
#pragma unroll
  for (int i = 0; i < 4; ++i) oacc[i] = vzero8();
  float mrow[8], lrow[8];
#pragma unroll
  for (int v = 0; v < 8; ++v) { mrow[v] = -1e30f; lrow[v] = 0.0f; }

  for (int kt = 0; kt < Nk; kt += 64) {
    __syncthreads();
    // K tile (row-major): async copy straight into LDS
    for (int u = tid; u < 64 * 8; u += 128) {
      int row = u >> 3, c = (u & 7) * 8;
      cp16_lds(&Kbh[(size_t)(kt + row) * DHEAD + c], &Ks[row * 72 + c]);
    }
    // V tile transposed into Vs[d][key]
    for (int u = tid; u < 64 * 64; u += 128) {
      int key = u >> 6, d = u & 63;
      Vs[d * 72 + key] = Vbh[(size_t)(kt + key) * DHEAD + d];
    }
    async_fence();
    __syncthreads();

    // S = Q @ K^T : 16 rows x 64 keys per wave
    v8f s[4];
#pragma unroll
    for (int nt = 0; nt < 4; ++nt) {
      const __bf16* kr = &Ks[(nt * 16 + nloc) * 72];
      v16bf b0 = frag_contig(kr + kh);
      v16bf b1 = frag_contig(kr + 32 + kh);
      s[nt] = vzero8();
      s[nt] = wmma_bf16(qa0, b0, s[nt]);
      s[nt] = wmma_bf16(qa1, b1, s[nt]);
    }

    // online softmax (rows striped across lane halves; reduce over 16 lanes)
    float pv[4][8];
#pragma unroll
    for (int v = 0; v < 8; ++v) {
      float tmax = fmaxf(fmaxf(s[0][v], s[1][v]), fmaxf(s[2][v], s[3][v]));
#pragma unroll
      for (int off = 1; off < 16; off <<= 1)
        tmax = fmaxf(tmax, __shfl_xor(tmax, off, 32));
      float mnew = fmaxf(mrow[v], tmax);
      float corr = __expf(mrow[v] - mnew);
      mrow[v] = mnew;
      float tsum = 0.0f;
#pragma unroll
      for (int nt = 0; nt < 4; ++nt) {
        float p = __expf(s[nt][v] - mnew);
        pv[nt][v] = p;
        tsum += p;
      }
#pragma unroll
      for (int off = 1; off < 16; off <<= 1)
        tsum += __shfl_xor(tsum, off, 32);
      lrow[v] = lrow[v] * corr + tsum;
#pragma unroll
      for (int dt = 0; dt < 4; ++dt) oacc[dt][v] *= corr;
    }

    // C-layout -> A-layout re-stage of P through per-wave LDS
#pragma unroll
    for (int nt = 0; nt < 4; ++nt)
#pragma unroll
      for (int v = 0; v < 8; ++v)
        Ps[wv][(mb + v) * 72 + nt * 16 + nloc] = (__bf16)pv[nt][v];
    __syncthreads();

    v16bf pa0, pa1;
    {
      const __bf16* pr = &Ps[wv][nloc * 72];
      int ka = (lane >> 4) * 8;
      pa0 = frag_split(pr + ka, pr + ka + 16);
      pa1 = frag_split(pr + 32 + ka, pr + 32 + ka + 16);
    }
#pragma unroll
    for (int dt = 0; dt < 4; ++dt) {
      const __bf16* vr = &Vs[(dt * 16 + nloc) * 72];
      v16bf b0 = frag_contig(vr + kh);
      v16bf b1 = frag_contig(vr + 32 + kh);
      oacc[dt] = wmma_bf16(pa0, b0, oacc[dt]);
      oacc[dt] = wmma_bf16(pa1, b1, oacc[dt]);
    }
  }

  // epilogue: normalize, write merged bf16 [B, Nq, 256]
#pragma unroll
  for (int dt = 0; dt < 4; ++dt) {
    int col = h * DHEAD + dt * 16 + nloc;
#pragma unroll
    for (int v = 0; v < 8; ++v) {
      int r = qTile * 64 + wv * 16 + mb + v;
      outM[((size_t)b * Nq + r) * 256 + col] = (__bf16)(oacc[dt][v] / lrow[v]);
    }
  }
}

// ---------------- LayerNorm + exact GELU: fp32 in -> bf16 out, 512 cols --------

__global__ __launch_bounds__(128)
void ln_gelu(const float* __restrict__ hin, __bf16* __restrict__ gout,
             const float* __restrict__ gamma, const float* __restrict__ beta,
             int C) {
  __shared__ float part[4];
  const int tid = threadIdx.x;
  const float* p = hin + (size_t)blockIdx.x * C;
  __bf16* q = gout + (size_t)blockIdx.x * C;
  float x[4];
#pragma unroll
  for (int i = 0; i < 4; ++i) x[i] = p[tid + i * 128];

  float s = x[0] + x[1] + x[2] + x[3];
#pragma unroll
  for (int off = 1; off < 32; off <<= 1) s += __shfl_xor(s, off, 32);
  if ((tid & 31) == 0) part[tid >> 5] = s;
  __syncthreads();
  float mean = (part[0] + part[1] + part[2] + part[3]) / (float)C;
  __syncthreads();

  float sv = 0.0f;
#pragma unroll
  for (int i = 0; i < 4; ++i) { float d = x[i] - mean; sv += d * d; }
#pragma unroll
  for (int off = 1; off < 32; off <<= 1) sv += __shfl_xor(sv, off, 32);
  if ((tid & 31) == 0) part[tid >> 5] = sv;
  __syncthreads();
  float var = (part[0] + part[1] + part[2] + part[3]) / (float)C;
  float inv = rsqrtf(var + 1e-5f);

#pragma unroll
  for (int i = 0; i < 4; ++i) {
    int col = tid + i * 128;
    float t = (x[i] - mean) * inv * gamma[col] + beta[col];
    q[col] = (__bf16)(0.5f * t * (1.0f + erff(t * 0.70710678118654752f)));
  }
}

// ---------------- launch ----------------

extern "C" void kernel_launch(void* const* d_in, const int* in_sizes, int n_in,
                              void* d_out, int out_size, void* d_ws, size_t ws_size,
                              hipStream_t stream) {
  (void)in_sizes; (void)n_in; (void)out_size; (void)ws_size;
  const float* x0  = (const float*)d_in[0];
  const float* x1  = (const float*)d_in[1];
  const float* Wqk = (const float*)d_in[2];
  const float* bqk = (const float*)d_in[3];
  const float* Wv  = (const float*)d_in[4];
  const float* bv  = (const float*)d_in[5];
  const float* Wp  = (const float*)d_in[6];
  const float* bp  = (const float*)d_in[7];
  const float* W1  = (const float*)d_in[8];
  const float* b1  = (const float*)d_in[9];
  const float* gm  = (const float*)d_in[10];
  const float* bt  = (const float*)d_in[11];
  const float* W2  = (const float*)d_in[12];
  const float* b2  = (const float*)d_in[13];

  const int R = RTOT;
  const float qscale = 0.35355339059327378f;  // 64^-0.25

  const size_t MB = 1024 * 1024, KB = 1024;
  char* ws = (char*)d_ws;
  __bf16* x0b  = (__bf16*)(ws + 0 * MB);          // 4MB
  __bf16* x1b  = (__bf16*)(ws + 4 * MB);          // 4MB
  __bf16* Wqkb = (__bf16*)(ws + 8 * MB);          // 128KB
  __bf16* Wvb  = (__bf16*)(ws + 8 * MB + 128 * KB);
  __bf16* Wpb  = (__bf16*)(ws + 8 * MB + 256 * KB);
  __bf16* W1b  = (__bf16*)(ws + 8 * MB + 384 * KB);  // 512KB
  __bf16* W2b  = (__bf16*)(ws + 8 * MB + 896 * KB);  // 256KB
  __bf16* qk0  = (__bf16*)(ws + 10 * MB);         // 4MB each
  __bf16* qk1  = (__bf16*)(ws + 14 * MB);
  __bf16* v0b  = (__bf16*)(ws + 18 * MB);
  __bf16* v1b  = (__bf16*)(ws + 22 * MB);
  __bf16* m0   = (__bf16*)(ws + 26 * MB);         // 4MB each
  __bf16* m1   = (__bf16*)(ws + 30 * MB);
  __bf16* p0   = (__bf16*)(ws + 34 * MB);
  __bf16* p1   = (__bf16*)(ws + 38 * MB);
  float*  h0   = (float*)(ws + 10 * MB);          // 16MB, reuse qk/v (dead)
  float*  h1   = (float*)(ws + 42 * MB);          // 16MB
  __bf16* g0   = (__bf16*)(ws + 26 * MB);         // 8MB, reuse m (dead)
  __bf16* g1   = (__bf16*)(ws + 34 * MB);         // 8MB, reuse p (dead)

  float* y0 = (float*)d_out;
  float* y1 = y0 + (size_t)R * 256;

  // ---- bulk fp32->bf16 conversions (once) ----
  cvt_f32_bf16<<<dim3(R * 256 / 1024), 128, 0, stream>>>(x0, x0b);
  cvt_f32_bf16<<<dim3(R * 256 / 1024), 128, 0, stream>>>(x1, x1b);
  cvt_f32_bf16<<<dim3(64), 128, 0, stream>>>(Wqk, Wqkb);
  cvt_f32_bf16<<<dim3(64), 128, 0, stream>>>(Wv, Wvb);
  cvt_f32_bf16<<<dim3(64), 128, 0, stream>>>(Wp, Wpb);
  cvt_f32_bf16<<<dim3(256), 128, 0, stream>>>(W1, W1b);
  cvt_f32_bf16<<<dim3(128), 128, 0, stream>>>(W2, W2b);

  dim3 g256(R / 64, 4), g512(R / 64, 8);

  // projections -> bf16 head layout [B,H,N,DH]
  gemm_bf16_wmma<<<g256, 128, 0, stream>>>(x0b, x0b, 256, Wqkb, bqk, nullptr,
      nullptr, qk0, 1, 256, 256, qscale);
  gemm_bf16_wmma<<<g256, 128, 0, stream>>>(x1b, x1b, 256, Wqkb, bqk, nullptr,
      nullptr, qk1, 1, 256, 256, qscale);
  gemm_bf16_wmma<<<g256, 128, 0, stream>>>(x0b, x0b, 256, Wvb, bv, nullptr,
      nullptr, v0b, 1, 256, 256, 1.0f);
  gemm_bf16_wmma<<<g256, 128, 0, stream>>>(x1b, x1b, 256, Wvb, bv, nullptr,
      nullptr, v1b, 1, 256, 256, 1.0f);

  // cross attention, both directions (merged bf16 output)
  dim3 ga(NSEQ / 64, NHEAD, 4);
  attn_wmma<<<ga, 128, 0, stream>>>(qk0, qk1, v1b, m0, NSEQ, NSEQ);
  attn_wmma<<<ga, 128, 0, stream>>>(qk1, qk0, v0b, m1, NSEQ, NSEQ);

  // Wp projection -> plain bf16
  gemm_bf16_wmma<<<g256, 128, 0, stream>>>(m0, m0, 256, Wpb, bp, nullptr,
      nullptr, p0, 0, 256, 256, 1.0f);
  gemm_bf16_wmma<<<g256, 128, 0, stream>>>(m1, m1, 256, Wpb, bp, nullptr,
      nullptr, p1, 0, 256, 256, 1.0f);

  // W1 on concat([x, p]) -> fp32 h
  gemm_bf16_wmma<<<g512, 128, 0, stream>>>(x0b, p0, 256, W1b, b1, nullptr,
      h0, nullptr, 0, 512, 512, 1.0f);
  gemm_bf16_wmma<<<g512, 128, 0, stream>>>(x1b, p1, 256, W1b, b1, nullptr,
      h1, nullptr, 0, 512, 512, 1.0f);

  // LayerNorm + GELU -> bf16 g
  ln_gelu<<<dim3(R), 128, 0, stream>>>(h0, g0, gm, bt, 512);
  ln_gelu<<<dim3(R), 128, 0, stream>>>(h1, g1, gm, bt, 512);

  // W2 + residual -> fp32 outputs
  gemm_bf16_wmma<<<g256, 128, 0, stream>>>(g0, g0, 512, W2b, b2, x0,
      y0, nullptr, 0, 512, 256, 1.0f);
  gemm_bf16_wmma<<<g256, 128, 0, stream>>>(g1, g1, 512, W2b, b2, x1,
      y1, nullptr, 0, 512, 256, 1.0f);
}
